// Segmentation_Head_45990509805625
// MI455X (gfx1250) — matrix operand order repaired and checked by reading, not compiled, not run on hardware
//
#include <hip/hip_runtime.h>
#include <hip/hip_bf16.h>

// ---------------------------------------------------------------------------
// CDNA5 (gfx1250) segmentation head.  wave32, WMMA f16 (f32 accumulate),
// double-buffered LDS staging, 64-wide K-steps (2 x v_wmma 16x16x32 per step).
// ---------------------------------------------------------------------------

typedef __attribute__((ext_vector_type(16))) _Float16 v16h;
typedef __attribute__((ext_vector_type(8)))  _Float16 v8h;
typedef __attribute__((ext_vector_type(8)))  float    v8f;

union Frag {
    v16h v;
    v8h  h8[2];
    _Float16 h[16];
};

#define LDA 72   // halfs per LDS row (144B -> every fragment chunk 16B aligned)
#define LDB 72
#define EPSBN 1e-5f

// A fragment (16x32 f16): lanes 0-15 row M=lane, K 0..7 -> e0..7, K 16..23 ->
// e8..15; lanes 16-31 shift K by +8.  Two ds_load_b128.
__device__ inline v16h load_frag_a(const _Float16* __restrict__ As, int row,
                                   int lh, int kOff) {
    Frag f;
    const _Float16* p = As + row * LDA + kOff + lh * 8;
    f.h8[0] = *(const v8h*)p;
    f.h8[1] = *(const v8h*)(p + 16);
    return f.v;
}
// B fragment (32x16 f16, stored transposed [n][k]): lanes 0-15 col N=lane,
// K 0..7 -> e0..7, K 16..23 -> e8..15; lanes 16-31 shift K by +8.  Two ds_load_b128.
__device__ inline v16h load_frag_b(const _Float16* __restrict__ Bs, int row,
                                   int lh, int kOff) {
    Frag f;
    const _Float16* p = Bs + row * LDB + kOff + lh * 8;
    f.h8[0] = *(const v8h*)p;
    f.h8[1] = *(const v8h*)(p + 16);
    return f.v;
}

// ---------------------------------------------------------------------------
// Implicit-GEMM conv (NCHW, 1x1 or 3x3 pad=1), BN+ReLU+residual epilogue,
// implicit channel concat via src1.  M tile 128 px (2 rows), N tile 64 co.
// 8 waves as 4(M) x 2(N); each wave 32x32 out via 2x2 WMMA, K-step 64.
// ---------------------------------------------------------------------------
__global__ __launch_bounds__(256) void conv_wmma_kernel(
    const float* __restrict__ src0, const float* __restrict__ src1,
    int ci0, int ci_total, int co_total,
    const float* __restrict__ wgt, int ks,
    const float* __restrict__ bn_g, const float* __restrict__ bn_b,
    int do_relu, const float* __restrict__ resid,
    float* __restrict__ out)
{
    __shared__ __align__(16) _Float16 AsBuf[2 * 128 * LDA];
    __shared__ __align__(16) _Float16 BsBuf[2 * 64 * LDB];

    const int tid   = threadIdx.x;
    const int lane  = tid & 31;
    const int wid   = tid >> 5;
    const int mW    = (wid & 3) * 32;
    const int nW    = (wid >> 2) * 32;
    const int lr    = lane & 15;
    const int lh    = lane >> 4;

    const int mBase = blockIdx.x * 128;
    const int nBase = blockIdx.y * 64;
    const int b     = mBase >> 12;        // 4096 px/image, tiles never straddle
    const int hwB   = mBase & 4095;
    const int h0    = hwB >> 6;
    const int ks2   = ks * ks;
    const int ci1   = ci_total - ci0;
    const int cpb   = ci_total >> 6;      // 64-wide ci chunks per tap
    const int S     = ks2 * cpb;          // pipeline stages

    v8f acc[2][2] = {};

    auto stage = [&](int s, int buf) {
        const int tap = s / cpb;
        const int cb  = (s - tap * cpb) << 6;
        const int dh  = (ks == 3) ? (tap / 3 - 1) : 0;
        const int dw  = (ks == 3) ? (tap % 3 - 1) : 0;
        _Float16* As = AsBuf + buf * (128 * LDA);
        _Float16* Bs = BsBuf + buf * (64 * LDB);
        {   // A tile: 128 px x 64 ci, im2col on the fly, f32 -> f16
            const int ciL = tid >> 2;            // 0..63
            const int pb  = (tid & 3) << 5;      // 32 consecutive pixels
            const int ci  = cb + ciL;
            const float* basep = (ci < ci0)
                ? src0 + ((size_t)b * ci0 + ci) * 4096
                : src1 + ((size_t)b * ci1 + (ci - ci0)) * 4096;
#pragma unroll
            for (int j = 0; j < 32; ++j) {
                const int m  = pb + j;
                const int hh = h0 + (m >> 6) + dh;
                const int ww = (m & 63) + dw;
                float v = 0.f;
                if ((unsigned)hh < 64u && (unsigned)ww < 64u)
                    v = basep[hh * 64 + ww];
                As[m * LDA + ciL] = (_Float16)v;
            }
        }
        {   // B tile: 64 co x 64 ci, weights OIHW (stride ks2 over ci)
            const int coL = tid >> 2;            // 0..63
            const int kg  = (tid & 3) << 4;      // 16 ci each
            const int co  = nBase + coL;
            const bool ok = (co < co_total);
#pragma unroll
            for (int j = 0; j < 16; ++j) {
                const int ci = cb + kg + j;
                float v = ok ? wgt[((size_t)co * ci_total + ci) * ks2 + tap] : 0.f;
                Bs[coL * LDB + kg + j] = (_Float16)v;
            }
            if (ok && cb + 64 < ci_total)        // -> global_prefetch_b8
                __builtin_prefetch(wgt + ((size_t)co * ci_total + cb + 64) * ks2 + tap, 0, 0);
        }
    };

    auto compute = [&](int buf) {
        const _Float16* As = AsBuf + buf * (128 * LDA);
        const _Float16* Bs = BsBuf + buf * (64 * LDB);
#pragma unroll
        for (int sub = 0; sub < 64; sub += 32) {
            const v16h a0 = load_frag_a(As, mW + lr,      lh, sub);
            const v16h a1 = load_frag_a(As, mW + 16 + lr, lh, sub);
            const v16h b0 = load_frag_b(Bs, nW + lr,      lh, sub);
            const v16h b1 = load_frag_b(Bs, nW + 16 + lr, lh, sub);
            acc[0][0] = __builtin_amdgcn_wmma_f32_16x16x32_f16(false, a0, false, b0, (short)0, acc[0][0], false, false);
            acc[0][1] = __builtin_amdgcn_wmma_f32_16x16x32_f16(false, a0, false, b1, (short)0, acc[0][1], false, false);
            acc[1][0] = __builtin_amdgcn_wmma_f32_16x16x32_f16(false, a1, false, b0, (short)0, acc[1][0], false, false);
            acc[1][1] = __builtin_amdgcn_wmma_f32_16x16x32_f16(false, a1, false, b1, (short)0, acc[1][1], false, false);
        }
    };

    stage(0, 0);
    __syncthreads();
    for (int s = 0; s < S; ++s) {
        if (s + 1 < S) stage(s + 1, (s + 1) & 1);   // fill other buffer
        compute(s & 1);
        __syncthreads();
    }

    // ---- epilogue: BN (eval, running var=1) + ReLU + residual
#pragma unroll
    for (int mi = 0; mi < 2; ++mi)
#pragma unroll
        for (int ni = 0; ni < 2; ++ni) {
            const int co = nBase + nW + ni * 16 + lr;
            if (co >= co_total) continue;
            float scale = 1.f, bias = 0.f;
            if (bn_g) { scale = bn_g[co] * rsqrtf(1.0f + EPSBN); bias = bn_b[co]; }
#pragma unroll
            for (int r = 0; r < 8; ++r) {
                const int mLoc = mW + mi * 16 + lh * 8 + r;
                const int hwg  = hwB + mLoc;
                float v = acc[mi][ni][r];
                if (bn_g)    v = v * scale + bias;
                if (do_relu) v = fmaxf(v, 0.f);
                const size_t o = ((size_t)(b * co_total + co)) * 4096 + hwg;
                if (resid) v += resid[o];
                out[o] = v;
            }
        }
}

// ---------------------------------------------------------------------------
// Generic WMMA GEMM: out[M,N] = A[M,K] * W[N,K]^T (+bias).  K % 64 == 0.
// Same tiling / double-buffered pipeline as the conv kernel.
// ---------------------------------------------------------------------------
__global__ __launch_bounds__(256) void gemm_wmma_kernel(
    const float* __restrict__ A, const float* __restrict__ W,
    const float* __restrict__ bias, float* __restrict__ out,
    int M, int N, int K)
{
    __shared__ __align__(16) _Float16 AsBuf[2 * 128 * LDA];
    __shared__ __align__(16) _Float16 BsBuf[2 * 64 * LDB];

    const int tid  = threadIdx.x;
    const int lane = tid & 31;
    const int wid  = tid >> 5;
    const int mW   = (wid & 3) * 32;
    const int nW   = (wid >> 2) * 32;
    const int lr   = lane & 15;
    const int lh   = lane >> 4;
    const int mBase = blockIdx.x * 128;
    const int nBase = blockIdx.y * 64;
    const int S     = K >> 6;

    v8f acc[2][2] = {};

    auto stage = [&](int s, int buf) {
        const int cb = s << 6;
        _Float16* As = AsBuf + buf * (128 * LDA);
        _Float16* Bs = BsBuf + buf * (64 * LDB);
        {   // A tile: 128 rows x 64 k (k contiguous in memory)
            const int mL = tid >> 1;
            const int kg = (tid & 1) << 5;
            const int m  = mBase + mL;
            const float* ap = A + (size_t)m * K + cb + kg;
#pragma unroll
            for (int j = 0; j < 32; ++j)
                As[mL * LDA + kg + j] = (_Float16)((m < M) ? ap[j] : 0.f);
        }
        {   // B tile: 64 n x 64 k (row-major weight, k contiguous)
            const int nL = tid >> 2;
            const int kg = (tid & 3) << 4;
            const int n  = nBase + nL;
            const float* wp = W + (size_t)n * K + cb + kg;
#pragma unroll
            for (int j = 0; j < 16; ++j)
                Bs[nL * LDB + kg + j] = (_Float16)((n < N) ? wp[j] : 0.f);
            if (n < N && cb + 64 < K)
                __builtin_prefetch(wp + 64, 0, 0);
        }
    };

    auto compute = [&](int buf) {
        const _Float16* As = AsBuf + buf * (128 * LDA);
        const _Float16* Bs = BsBuf + buf * (64 * LDB);
#pragma unroll
        for (int sub = 0; sub < 64; sub += 32) {
            const v16h a0 = load_frag_a(As, mW + lr,      lh, sub);
            const v16h a1 = load_frag_a(As, mW + 16 + lr, lh, sub);
            const v16h b0 = load_frag_b(Bs, nW + lr,      lh, sub);
            const v16h b1 = load_frag_b(Bs, nW + 16 + lr, lh, sub);
            acc[0][0] = __builtin_amdgcn_wmma_f32_16x16x32_f16(false, a0, false, b0, (short)0, acc[0][0], false, false);
            acc[0][1] = __builtin_amdgcn_wmma_f32_16x16x32_f16(false, a0, false, b1, (short)0, acc[0][1], false, false);
            acc[1][0] = __builtin_amdgcn_wmma_f32_16x16x32_f16(false, a1, false, b0, (short)0, acc[1][0], false, false);
            acc[1][1] = __builtin_amdgcn_wmma_f32_16x16x32_f16(false, a1, false, b1, (short)0, acc[1][1], false, false);
        }
    };

    stage(0, 0);
    __syncthreads();
    for (int s = 0; s < S; ++s) {
        if (s + 1 < S) stage(s + 1, (s + 1) & 1);
        compute(s & 1);
        __syncthreads();
    }

#pragma unroll
    for (int mi = 0; mi < 2; ++mi)
#pragma unroll
        for (int ni = 0; ni < 2; ++ni) {
            const int n = nBase + nW + ni * 16 + lr;
            if (n >= N) continue;
            const float bv = bias ? bias[n] : 0.f;
#pragma unroll
            for (int r = 0; r < 8; ++r) {
                const int m = mBase + mW + mi * 16 + lh * 8 + r;
                if (m >= M) continue;
                out[(size_t)m * N + n] = acc[mi][ni][r] + bv;
            }
        }
}

// ---------------------------------------------------------------------------
// Stage 1: per (b,p): pix = softmax_n(I_p), F_l[k,c] = R * sum_n pix*F_p,
// plus materialize F_p rows ((b*P+p)*64+n) x 512 for the query GEMM.
// ---------------------------------------------------------------------------
__global__ __launch_bounds__(256) void patch_fl_kernel(
    const float* __restrict__ F, const float* __restrict__ I,
    const float* __restrict__ R, float* __restrict__ Fp, float* __restrict__ Fl)
{
    __shared__ float pix[64 * 20];
    const int tid = threadIdx.x;
    const int bp  = blockIdx.x;
    const int b   = bp >> 6;
    const int p   = bp & 63;
    const int base = (p >> 3) * 512 + (p & 7) * 8;   // hw of patch origin

    for (int idx = tid; idx < 19 * 64; idx += 256) {
        const int k = idx >> 6, n = idx & 63;
        pix[n * 20 + k] = I[((size_t)(b * 19 + k)) * 4096 + base + ((n >> 3) << 6) + (n & 7)];
    }
    __syncthreads();
    if (tid < 19) {
        const int k = tid;
        float mx = -3.4e38f;
        for (int n = 0; n < 64; ++n) mx = fmaxf(mx, pix[n * 20 + k]);
        float s = 0.f;
        for (int n = 0; n < 64; ++n) { float e = expf(pix[n * 20 + k] - mx); pix[n * 20 + k] = e; s += e; }
        const float inv = 1.f / s;
        for (int n = 0; n < 64; ++n) pix[n * 20 + k] *= inv;
    }
    __syncthreads();

    for (int idx = tid; idx < 64 * 512; idx += 256) {      // F_p rows
        const int c = idx & 511, n = idx >> 9;
        Fp[((size_t)(bp * 64 + n)) * 512 + c] =
            F[((size_t)(b * 512 + c)) * 4096 + base + ((n >> 3) << 6) + (n & 7)];
    }
    for (int idx = tid; idx < 19 * 512; idx += 256) {      // F_l[k][c]
        const int c = idx & 511, k = idx >> 9;
        const float* fc = F + ((size_t)(b * 512 + c)) * 4096 + base;
        float s = 0.f;
        for (int n = 0; n < 64; ++n)
            s += pix[n * 20 + k] * fc[((n >> 3) << 6) + (n & 7)];
        Fl[((size_t)(bp * 19 + k)) * 512 + c] = s * R[((b * 19 + k) << 6) + p];
    }
}

// h = relu(lg_w1 @ F_l + F_l)  (mix over the 64 patches)
__global__ __launch_bounds__(256) void lg1_kernel(
    const float* __restrict__ Fl, const float* __restrict__ w1,
    float* __restrict__ Fh, int total)
{
    const int idx = blockIdx.x * 256 + threadIdx.x;
    if (idx >= total) return;
    const int c  = idx & 511;
    const int r  = idx >> 9;
    const int k  = r % 19;
    const int r2 = r / 19;
    const int q  = r2 & 63;
    const int b  = r2 >> 6;
    const float* flb = Fl + ((size_t)(b * 64) * 19 + k) * 512 + c;
    float s = 0.f;
#pragma unroll 4
    for (int p = 0; p < 64; ++p) s += w1[q * 64 + p] * flb[(size_t)p * 9728];
    s += flb[(size_t)q * 9728];
    Fh[idx] = fmaxf(s, 0.f);
}

// F_g[b,k,c] = sum_p fuse_w[p] * F_l2[b,p,k,c] + fuse_b
__global__ __launch_bounds__(256) void fuse_kernel(
    const float* __restrict__ Fl2, const float* __restrict__ fw,
    const float* __restrict__ fbias, float* __restrict__ Fg)
{
    const int idx = blockIdx.x * 256 + threadIdx.x;
    if (idx >= 4 * 19 * 512) return;
    const int c = idx & 511;
    const int k = (idx >> 9) % 19;
    const int b = (idx >> 9) / 19;
    const float* fb2 = Fl2 + ((size_t)(b * 64) * 19 + k) * 512 + c;
    float s = fbias[0];
#pragma unroll 4
    for (int p = 0; p < 64; ++p) s += fw[p] * fb2[(size_t)p * 9728];
    Fg[idx] = s;
}

// per (b,p): aff = softmax_k(q . key^T); F_s = aff @ value, written NCHW
__global__ __launch_bounds__(256) void attn_kernel(
    const float* __restrict__ query, const float* __restrict__ key,
    const float* __restrict__ value, float* __restrict__ Fs)
{
    __shared__ float ksh[19 * 256];
    __shared__ float vsh[19 * 256];
    __shared__ float aff[64 * 20];
    const int tid = threadIdx.x;
    const int bp  = blockIdx.x;
    const int b   = bp >> 6;
    const int p   = bp & 63;

    for (int idx = tid; idx < 19 * 256; idx += 256) {
        ksh[idx] = key[(size_t)bp * 19 * 256 + idx];
        vsh[idx] = value[(size_t)b * 19 * 256 + idx];
    }
    __syncthreads();
    for (int idx = tid; idx < 64 * 19; idx += 256) {
        const int n = idx / 19, k = idx % 19;
        const float* qr = query + ((size_t)bp * 64 + n) * 256;
        const float* kr = ksh + k * 256;
        float s = 0.f;
#pragma unroll 4
        for (int d = 0; d < 256; ++d) s += qr[d] * kr[d];
        aff[n * 20 + k] = s;
    }
    __syncthreads();
    if (tid < 64) {
        float* row = aff + tid * 20;
        float mx = -3.4e38f;
        for (int k = 0; k < 19; ++k) mx = fmaxf(mx, row[k]);
        float s = 0.f;
        for (int k = 0; k < 19; ++k) { float e = expf(row[k] - mx); row[k] = e; s += e; }
        const float inv = 1.f / s;
        for (int k = 0; k < 19; ++k) row[k] *= inv;
    }
    __syncthreads();
    const int base = (p >> 3) * 512 + (p & 7) * 8;
    for (int idx = tid; idx < 64 * 256; idx += 256) {
        const int d = idx & 255, n = idx >> 8;
        float s = 0.f;
#pragma unroll
        for (int k = 0; k < 19; ++k) s += aff[n * 20 + k] * vsh[k * 256 + d];
        Fs[((size_t)(b * 256 + d)) * 4096 + base + ((n >> 3) << 6) + (n & 7)] = s;
    }
}

// ---------------------------------------------------------------------------
extern "C" void kernel_launch(void* const* d_in, const int* in_sizes, int n_in,
                              void* d_out, int out_size, void* d_ws, size_t ws_size,
                              hipStream_t stream) {
    (void)in_sizes; (void)n_in; (void)out_size; (void)ws_size;
    const float* M_1    = (const float*)d_in[0];
    const float* F      = (const float*)d_in[1];
    const float* I      = (const float*)d_in[2];
    const float* R      = (const float*)d_in[3];
    const float* lg_w1  = (const float*)d_in[4];
    const float* lg_w2  = (const float*)d_in[5];
    const float* fuse_w = (const float*)d_in[6];
    const float* fuse_b = (const float*)d_in[7];
    const float* q_w    = (const float*)d_in[8];
    const float* q_b    = (const float*)d_in[9];
    const float* k_w    = (const float*)d_in[10];
    const float* k_b    = (const float*)d_in[11];
    const float* v_w    = (const float*)d_in[12];
    const float* v_b    = (const float*)d_in[13];
    const float* c1_w   = (const float*)d_in[14];
    const float* bn1_g  = (const float*)d_in[15];
    const float* bn1_b  = (const float*)d_in[16];
    const float* c2_w   = (const float*)d_in[17];
    const float* bn2_g  = (const float*)d_in[18];
    const float* bn2_b  = (const float*)d_in[19];
    const float* c3_w   = (const float*)d_in[20];
    const float* bn3_g  = (const float*)d_in[21];
    const float* bn3_b  = (const float*)d_in[22];
    const float* drop_w = (const float*)d_in[23];
    float* out = (float*)d_out;
    float* w   = (float*)d_ws;

    // workspace layout (floats); x2 aliases Fp, x3 aliases query+Fs
    float* Fp   = w;                    // 8,388,608  (16384 x 512)
    float* Fl   = Fp  + 8388608ULL;     // 2,490,368  (4864 x 512)
    float* Fh   = Fl  + 2490368ULL;     // 2,490,368
    float* Fl2  = Fh  + 2490368ULL;     // 2,490,368
    float* Fg   = Fl2 + 2490368ULL;     //    38,912  (76 x 512)
    float* Key  = Fg  + 38912ULL;       // 1,245,184  (4864 x 256)
    float* Val  = Key + 1245184ULL;     //    19,456  (76 x 256)
    float* Q    = Val + 19456ULL;       // 4,194,304  (16384 x 256)
    float* Fs   = Q   + 4194304ULL;     // 4,194,304  (4 x 256 x 4096)
    float* Fo1  = Fs  + 4194304ULL;     // 8,388,608  (4 x 512 x 4096)
    float* x2   = Fp;                   // Fp dead after query GEMM
    float* x3   = Q;                    // Q+Fs dead after conv1

    // 1) pixel softmax + F_l + F_p materialization
    patch_fl_kernel<<<256, 256, 0, stream>>>(F, I, R, Fp, Fl);
    // 2) LG mixing over patches
    lg1_kernel<<<(2490368 + 255) / 256, 256, 0, stream>>>(Fl, lg_w1, Fh, 2490368);
    // 3) F_l2 = h @ lg_w2^T    (4864 x 512 x 512)
    gemm_wmma_kernel<<<dim3(38, 8), 256, 0, stream>>>(Fh, lg_w2, nullptr, Fl2, 4864, 512, 512);
    // 4) fuse -> F_g
    fuse_kernel<<<(38912 + 255) / 256, 256, 0, stream>>>(Fl2, fuse_w, fuse_b, Fg);
    // 5) key / value / query projections
    gemm_wmma_kernel<<<dim3(38, 4), 256, 0, stream>>>(Fl2, k_w, k_b, Key, 4864, 256, 512);
    gemm_wmma_kernel<<<dim3(1, 4), 256, 0, stream>>>(Fg, v_w, v_b, Val, 76, 256, 512);
    gemm_wmma_kernel<<<dim3(128, 4), 256, 0, stream>>>(Fp, q_w, q_b, Q, 16384, 256, 512);
    // 6) attention + patch_recover -> Fs (NCHW)
    attn_kernel<<<256, 256, 0, stream>>>(Q, Key, Val, Fs);
    // 7) Fo1 = F + bn_relu(conv1x1(Fs))
    conv_wmma_kernel<<<dim3(128, 8), 256, 0, stream>>>(
        Fs, nullptr, 256, 256, 512, c1_w, 1, bn1_g, bn1_b, 1, F, Fo1);
    // 8) x2 = bn_relu(conv3x3(Fo1))
    conv_wmma_kernel<<<dim3(128, 8), 256, 0, stream>>>(
        Fo1, nullptr, 512, 512, 512, c2_w, 3, bn2_g, bn2_b, 1, nullptr, x2);
    // 9) x3 = bn_relu(conv3x3(concat(x2, M_1)))   (implicit concat)
    conv_wmma_kernel<<<dim3(128, 8), 256, 0, stream>>>(
        x2, M_1, 512, 2560, 512, c3_w, 3, bn3_g, bn3_b, 1, nullptr, x3);
    // 10) classifier 1x1 -> d_out (4 x 19 x 64 x 64)
    conv_wmma_kernel<<<dim3(128, 1), 256, 0, stream>>>(
        x3, nullptr, 512, 512, 19, drop_w, 1, nullptr, nullptr, 0, nullptr, out);
}
